// MultiHeadAttention_27951647163020
// MI455X (gfx1250) — matrix operand order repaired and checked by reading, not compiled, not run on hardware
//
#include <hip/hip_runtime.h>
#include <hip/hip_bf16.h>

// ---------------------------------------------------------------------------
// MHA forward for MI455X (gfx1250, wave32, WMMA).
// v6: v4 data path (f32->bf16 once; TDM tensor_load_to_lds stages the 64x1024
// B tile in LDS; WMMA bf16/f32-accum). GEMM k-loop reworked: 2 k-steps per
// iteration, all fragment loads grouped at the top, WMMAs after — pipelining
// via scheduler hoisting, no ping/pong register rotation (v5's v_dual_mov
// storm) and no conditional in the body (keeps the loop unrollable).
// ---------------------------------------------------------------------------

typedef __attribute__((ext_vector_type(16))) __bf16 bf16x16;
typedef __attribute__((ext_vector_type(8)))  float  f32x8;
typedef __attribute__((ext_vector_type(4)))  unsigned int u32x4;
typedef __attribute__((ext_vector_type(8)))  int  i32x8;
typedef __attribute__((ext_vector_type(4)))  int  i32x4;
typedef unsigned short ushort;

#define DIMC   1024
#define SEQ    1024
#define NB     4
#define NH     16
#define HD     64

#if __has_builtin(__builtin_amdgcn_tensor_load_to_lds)
#define HAVE_TDM 1
#else
#define HAVE_TDM 0
#endif

__device__ __forceinline__ ushort f2bf(float f) {
  unsigned u = __builtin_bit_cast(unsigned, f);
  u += 0x7fffu + ((u >> 16) & 1u);          // round-to-nearest-even
  return (ushort)(u >> 16);
}

// A fragment (16x32 bf16 MxK): lane<16 K={0..15}, lane>=16 +16 (folded
// into p by caller). Two 16B loads: p[0..7], p[8..15].
__device__ __forceinline__ bf16x16 bfrag_gap(const ushort* p) {
  union { bf16x16 v; uint4 q[2]; } u;
  u.q[0] = *(const uint4*)(p + 0);
  u.q[1] = *(const uint4*)(p + 8);
  return u.v;
}

// B fragment (32x16 bf16 KxN): lane = column N, 16 consecutive K (+16 for
// hi lanes folded into p). Two 16B loads.
__device__ __forceinline__ bf16x16 bfrag_contig(const ushort* p) {
  union { bf16x16 v; uint4 q[2]; } u;
  u.q[0] = *(const uint4*)(p + 0);
  u.q[1] = *(const uint4*)(p + 8);
  return u.v;
}

__device__ __forceinline__ f32x8 wmma_bf16(bf16x16 a, bf16x16 b, f32x8 c) {
  return __builtin_amdgcn_wmma_f32_16x16x32_bf16(false, a, false, b,
                                                 (short)0, c, false, false);
}

__device__ __forceinline__ void wait_tensorcnt0() {
#if __has_builtin(__builtin_amdgcn_s_wait_tensorcnt)
  __builtin_amdgcn_s_wait_tensorcnt(0);
#else
  asm volatile("s_wait_tensorcnt 0x0" ::: "memory");
#endif
}

// ---------------------------------------------------------------------------
// One-shot f32 -> packed bf16 (8 elements / thread; n % 8 == 0).
// ---------------------------------------------------------------------------
__global__ __launch_bounds__(256)
void cvt_f32_bf16(const float* __restrict__ in, ushort* __restrict__ out, int n) {
  int i = (blockIdx.x * 256 + threadIdx.x) * 8;
  if (i >= n) return;
  float4 a = *(const float4*)(in + i);
  float4 b = *(const float4*)(in + i + 4);
  uint4 o;
  o.x = (unsigned)f2bf(a.x) | ((unsigned)f2bf(a.y) << 16);
  o.y = (unsigned)f2bf(a.z) | ((unsigned)f2bf(a.w) << 16);
  o.z = (unsigned)f2bf(b.x) | ((unsigned)f2bf(b.y) << 16);
  o.w = (unsigned)f2bf(b.z) | ((unsigned)f2bf(b.w) << 16);
  *(uint4*)(out + i) = o;
}

// ---------------------------------------------------------------------------
// GEMM: out[m][n] = sum_k A[m][k] * Bm[n][k] + bias[n]   (A, Bm bf16)
// Block 128x64, 8 waves; B-tile (64x1024, 128KB) DMA'd to LDS by the TDM
// once per block. Loop body = 2 k-steps: 20 loads up front, 8 WMMAs after.
// mode 0: bf16 [b,h,s,d] | mode 1: bf16 [b,h,d,s] | mode 2: f32 [m][n]
// ---------------------------------------------------------------------------
#define GEMM_LDS_BYTES (64 * DIMC * 2)

__global__ __launch_bounds__(256)
void mha_gemm(const ushort* __restrict__ A, const ushort* __restrict__ Bm,
              const float* __restrict__ bias, void* __restrict__ out,
              int mode) {
  extern __shared__ ushort btile[];            // 64 x 1024 bf16

  const int tid  = threadIdx.x;
  const int w    = tid >> 5;
  const int lane = tid & 31;
  const int l16  = lane & 15;
  const int hiA  = (lane >> 4) ? 16 : 0;
  const int hiB  = (lane >> 4) ? 16 : 0;
  const int hi8  = (lane >> 4) ? 8 : 0;

  const int mBase = blockIdx.x * 128 + w * 16;
  const int nBase = blockIdx.y * 64;

  const ushort* bsrc = Bm + (size_t)nBase * DIMC;   // contiguous 64x1024 tile

#if HAVE_TDM
  // --- Tensor Data Mover: one 2D descriptor moves the whole B tile ---------
  if (w == 0) {
    const unsigned ldsa = (unsigned)(unsigned long long)(uintptr_t)btile;
    const unsigned long long ga = (unsigned long long)(uintptr_t)bsrc;
    // D# group0: count=1 | lds_addr | global_addr[56:0] | type=2 (bits 127:126)
    u32x4 g0 = { 1u, ldsa, (unsigned)ga,
                 (unsigned)((ga >> 32) & 0x1FFFFFFu) | (2u << 30) };
    // D# group1: data_size=2B | tensor_dim0=1024 | tensor_dim1=64 |
    //            tile_dim0=1024 | tile_dim1=64 | tensor_dim0_stride=1024
    i32x8 g1 = { (int)(1u << 16),
                 (int)(1024u << 16),
                 (int)(64u << 16),
                 (int)(1024u << 16),
                 64,
                 1024,
                 0, 0 };
    i32x4 gz4 = { 0, 0, 0, 0 };            // groups 2/3 unused (2D tensor)
    i32x8 gz8 = { 0, 0, 0, 0, 0, 0, 0, 0 };
    __builtin_amdgcn_tensor_load_to_lds(g0, g1, gz4, gz4, gz8, 0);
    wait_tensorcnt0();
  }
  __syncthreads();
#else
  // --- Fallback: cooperative vector copy ------------------------------------
  for (int j = 0; j < 32; ++j) {
    const int e = (j * 256 + tid) * 8;
    *(uint4*)(btile + e) = *(const uint4*)(bsrc + e);
  }
  __syncthreads();
#endif

  const ushort* aRow = A + (size_t)(mBase + l16) * DIMC + hiA;
  const ushort* bL[4];
#pragma unroll
  for (int nt = 0; nt < 4; ++nt)
    bL[nt] = btile + (nt * 16 + l16) * DIMC + hiB;

  f32x8 acc[4] = {};
  for (int k0 = 0; k0 < DIMC; k0 += 64) {
    __builtin_prefetch(aRow + k0 + 128, 0, 1);   // global_prefetch_b8
    // All fragment loads for both k-steps issued up front.
    bf16x16 a0 = bfrag_gap(aRow + k0);
    bf16x16 a1 = bfrag_gap(aRow + k0 + 32);
    bf16x16 b0[4], b1[4];
#pragma unroll
    for (int nt = 0; nt < 4; ++nt) b0[nt] = bfrag_contig(bL[nt] + k0);
#pragma unroll
    for (int nt = 0; nt < 4; ++nt) b1[nt] = bfrag_contig(bL[nt] + k0 + 32);
    // Matrix work.
#pragma unroll
    for (int nt = 0; nt < 4; ++nt) acc[nt] = wmma_bf16(a0, b0[nt], acc[nt]);
#pragma unroll
    for (int nt = 0; nt < 4; ++nt) acc[nt] = wmma_bf16(a1, b1[nt], acc[nt]);
  }

  // Epilogue: lane l16 = column, VGPR i (+hi8) = row.
#pragma unroll
  for (int nt = 0; nt < 4; ++nt) {
    const int n  = nBase + nt * 16 + l16;
    const float bv = bias[n];
#pragma unroll
    for (int i = 0; i < 8; ++i) {
      const int m = mBase + i + hi8;
      const float v = acc[nt][i] + bv;
      if (mode == 0) {          // [b,h,s,d] bf16
        ((ushort*)out)[((m >> 10) * NH + (n >> 6)) * (SEQ * HD)
                       + (m & 1023) * HD + (n & 63)] = f2bf(v);
      } else if (mode == 1) {   // [b,h,d,s] bf16
        ((ushort*)out)[((m >> 10) * NH + (n >> 6)) * (SEQ * HD)
                       + (n & 63) * SEQ + (m & 1023)] = f2bf(v);
      } else {                  // [m][n] f32
        ((float*)out)[m * DIMC + n] = v;
      }
    }
  }
}

// ---------------------------------------------------------------------------
// Fused attention per (b, h, 32-query tile):
//   scores(32x1024 f32, LDS) = 0.125 * Qh Kh^T + mask ; softmax^3 in LDS;
//   final pass writes bf16 attn tile ; out(32x64) = attn @ Vh -> bf16.
// 8 waves: msub = w&1 (16-row strip), nq = w>>1 (column group).
// ---------------------------------------------------------------------------
#define SC_FLOATS   (32 * SEQ)
#define RED_FLOATS  (32 * 8 + 32)
#define SMEM_BYTES  ((SC_FLOATS + RED_FLOATS) * 4 + 32 * SEQ * 2)

__global__ __launch_bounds__(256)
void mha_attn(const ushort* __restrict__ Qh, const ushort* __restrict__ Kh,
              const ushort* __restrict__ VhT, const float* __restrict__ mask,
              ushort* __restrict__ out) {
  extern __shared__ float smem[];
  float*  sc    = smem;                                     // 32x1024 f32
  float*  part  = smem + SC_FLOATS;                         // 32x8
  float*  rowv  = part + 32 * 8;                            // 32
  ushort* attnB = (ushort*)(smem + SC_FLOATS + RED_FLOATS); // 32x1024 bf16

  const int qt = blockIdx.x, h = blockIdx.y, b = blockIdx.z;
  const int tid  = threadIdx.x;
  const int w    = tid >> 5;
  const int lane = tid & 31;
  const int l16  = lane & 15;
  const int hiA  = (lane >> 4) ? 16 : 0;
  const int hiB  = (lane >> 4) ? 16 : 0;
  const int hi8  = (lane >> 4) ? 8 : 0;
  const int msub = w & 1, nq = w >> 1;

  const ushort* qbase = Qh  + (b * NH + h) * (SEQ * HD);  // [s][d]
  const ushort* kbase = Kh  + (b * NH + h) * (SEQ * HD);  // [s][d]
  const ushort* vtb   = VhT + (b * NH + h) * (SEQ * HD);  // [d][s]

  // --- Phase 1: scores = 0.125 * Q K^T + mask --------------------------------
  const int mRow = qt * 32 + msub * 16 + l16;
  const bf16x16 aq0 = bfrag_gap(qbase + mRow * HD + 0  + hiA);
  const bf16x16 aq1 = bfrag_gap(qbase + mRow * HD + 32 + hiA);

  for (int nt = nq; nt < SEQ / 16; nt += 4) {
    const int n = nt * 16 + l16;                 // key position
    const ushort* kr = kbase + n * HD;
    f32x8 c = {};
    c = wmma_bf16(aq0, bfrag_contig(kr + 0  + hiB), c);
    c = wmma_bf16(aq1, bfrag_contig(kr + 32 + hiB), c);
    const float mk = mask[b * SEQ + n];
#pragma unroll
    for (int i = 0; i < 8; ++i)
      sc[(msub * 16 + i + hi8) * SEQ + n] = c[i] * 0.125f + mk;
  }
  __syncthreads();

  // --- Phase 2: triple softmax; last pass emits bf16 -------------------------
  {
    const int r = tid >> 3, p = tid & 7;         // 8 threads per 1024-row
    float* row = sc + r * SEQ;
    const int c0 = p * 128, c1 = c0 + 128;
    for (int it = 0; it < 3; ++it) {
      float mx = -3.4e38f;
      for (int c = c0; c < c1; ++c) mx = fmaxf(mx, row[c]);
      part[r * 8 + p] = mx;
      __syncthreads();
      if (p == 0) {
        float m2 = part[r * 8];
        for (int j = 1; j < 8; ++j) m2 = fmaxf(m2, part[r * 8 + j]);
        rowv[r] = m2;
      }
      __syncthreads();
      const float rm = rowv[r];
      float s = 0.f;
      for (int c = c0; c < c1; ++c) {
        const float e = __expf(row[c] - rm);
        row[c] = e;
        s += e;
      }
      part[r * 8 + p] = s;
      __syncthreads();
      if (p == 0) {
        float s2 = 0.f;
        for (int j = 0; j < 8; ++j) s2 += part[r * 8 + j];
        rowv[r] = s2;
      }
      __syncthreads();
      const float inv = 1.f / rowv[r];
      if (it < 2) {
        for (int c = c0; c < c1; ++c) row[c] *= inv;
      } else {                       // final pass: normalized bf16 attn tile
        for (int c = c0; c < c1; ++c) attnB[r * SEQ + c] = f2bf(row[c] * inv);
      }
      __syncthreads();
    }
  }

  // --- Phase 3: out = attn @ Vh (A bf16 from LDS, B = VhT rows) --------------
  const ushort* v0   = vtb + (nq * 16 + l16) * SEQ + hiB;
  const ushort* arow = attnB + (msub * 16 + l16) * SEQ + hiA;
  f32x8 o = {};
  for (int kt = 0; kt < SEQ; kt += 64) {
    __builtin_prefetch(v0 + kt + 128, 0, 1);
    bf16x16 af0 = bfrag_gap(arow + kt);
    bf16x16 vf0 = bfrag_contig(v0 + kt);
    bf16x16 af1 = bfrag_gap(arow + kt + 32);
    bf16x16 vf1 = bfrag_contig(v0 + kt + 32);
    o = wmma_bf16(af0, vf0, o);
    o = wmma_bf16(af1, vf1, o);
  }

  // Store merged-head bf16 layout [b, s, h*64 + d] (A of the final GEMM).
  const int outBase = (b * SEQ + qt * 32) * DIMC + h * HD + nq * 16 + l16;
#pragma unroll
  for (int i = 0; i < 8; ++i)
    out[outBase + (msub * 16 + i + hi8) * DIMC] = f2bf(o[i]);
}

// ---------------------------------------------------------------------------
extern "C" void kernel_launch(void* const* d_in, const int* in_sizes, int n_in,
                              void* d_out, int out_size, void* d_ws, size_t ws_size,
                              hipStream_t stream) {
  (void)in_sizes; (void)n_in; (void)out_size; (void)ws_size;

  const float* q    = (const float*)d_in[0];
  const float* k    = (const float*)d_in[1];
  const float* v    = (const float*)d_in[2];
  const float* mask = (const float*)d_in[3];
  const float* Wq   = (const float*)d_in[4];
  const float* bq   = (const float*)d_in[5];
  const float* Wk   = (const float*)d_in[6];
  const float* bk   = (const float*)d_in[7];
  const float* Wv   = (const float*)d_in[8];
  const float* bv   = (const float*)d_in[9];
  const float* Wo   = (const float*)d_in[10];
  const float* bo   = (const float*)d_in[11];

  ushort* ws = (ushort*)d_ws;
  const size_t ACT = (size_t)NB * SEQ * DIMC;  // 4M elems
  const size_t WSZ = (size_t)DIMC * DIMC;      // 1M elems
  ushort* qb  = ws;
  ushort* kb  = qb  + ACT;
  ushort* vb  = kb  + ACT;
  ushort* Wqb = vb  + ACT;
  ushort* Wkb = Wqb + WSZ;
  ushort* Wvb = Wkb + WSZ;
  ushort* Wob = Wvb + WSZ;
  ushort* Qh  = Wob + WSZ;
  ushort* Kh  = Qh  + ACT;
  ushort* VhT = Kh  + ACT;
  ushort* AOut= VhT + ACT;

  (void)hipFuncSetAttribute((const void*)mha_gemm,
                            hipFuncAttributeMaxDynamicSharedMemorySize,
                            (int)GEMM_LDS_BYTES);
  (void)hipFuncSetAttribute((const void*)mha_attn,
                            hipFuncAttributeMaxDynamicSharedMemorySize,
                            (int)SMEM_BYTES);

  const dim3 blk(256);
  const int actBlocks = (int)(ACT / 8 / 256);   // 2048
  const int wBlocks   = (int)(WSZ / 8 / 256);   // 512
  cvt_f32_bf16<<<actBlocks, blk, 0, stream>>>(q,  qb,  (int)ACT);
  cvt_f32_bf16<<<actBlocks, blk, 0, stream>>>(k,  kb,  (int)ACT);
  cvt_f32_bf16<<<actBlocks, blk, 0, stream>>>(v,  vb,  (int)ACT);
  cvt_f32_bf16<<<wBlocks,   blk, 0, stream>>>(Wq, Wqb, (int)WSZ);
  cvt_f32_bf16<<<wBlocks,   blk, 0, stream>>>(Wk, Wkb, (int)WSZ);
  cvt_f32_bf16<<<wBlocks,   blk, 0, stream>>>(Wv, Wvb, (int)WSZ);
  cvt_f32_bf16<<<wBlocks,   blk, 0, stream>>>(Wo, Wob, (int)WSZ);

  const dim3 gGrid(32, 16);   // M/128 x N/64
  mha_gemm<<<gGrid, blk, GEMM_LDS_BYTES, stream>>>(qb, Wqb, bq, Qh,  0);
  mha_gemm<<<gGrid, blk, GEMM_LDS_BYTES, stream>>>(kb, Wkb, bk, Kh,  0);
  mha_gemm<<<gGrid, blk, GEMM_LDS_BYTES, stream>>>(vb, Wvb, bv, VhT, 1);

  const dim3 aGrid(SEQ / 32, NH, NB);  // 32 x 16 x 4
  mha_attn<<<aGrid, blk, SMEM_BYTES, stream>>>(Qh, Kh, VhT, mask, AOut);

  mha_gemm<<<gGrid, blk, GEMM_LDS_BYTES, stream>>>(AOut, Wob, bo, d_out, 2);
}
